// GraphSageSupervised_9603546873884
// MI455X (gfx1250) — compile-verified, matrix operand order checked by reading, not run on hardware
//
#include <hip/hip_runtime.h>
#include <hip/hip_bf16.h>
#include <stdint.h>

#define S_FAN 25
#define D_FEAT 128

typedef __attribute__((ext_vector_type(16))) __bf16 v16bf;
typedef __attribute__((ext_vector_type(8)))  __bf16 v8bf;
typedef __attribute__((ext_vector_type(8)))  float  v8f;

// ---------------------------------------------------------------------------
// Weight convert + transpose: Wt[n][k] = bf16(W[k][n]);  W is [256][128] f32.
// n-major bf16 makes each WMMA B-fragment K-group contiguous in memory.
// ---------------------------------------------------------------------------
__global__ void __launch_bounds__(256) wconv_kernel(const float* __restrict__ W,
                                                    __bf16* __restrict__ Wt) {
  int idx = blockIdx.x * 256 + threadIdx.x;  // 0 .. 256*128-1
  int k = idx >> 7;
  int n = idx & 127;
  Wt[n * 256 + k] = (__bf16)W[k * 128 + n];
}

// ---------------------------------------------------------------------------
// Mean over S sampled neighbors: out[r][c] = (1/S) * sum_s in[r*S+s][c]
// Threads span columns -> every one of the S row-reads is a fully coalesced
// 512B segment; the big 327MB tensors stream through HBM exactly once.
// ---------------------------------------------------------------------------
__global__ void __launch_bounds__(256) mean_kernel(const float* __restrict__ in,
                                                   float* __restrict__ out,
                                                   int nRows) {
  int idx = blockIdx.x * 256 + threadIdx.x;
  if (idx >= nRows * D_FEAT) return;
  int r = idx >> 7;
  int c = idx & 127;
  const float* p = in + (size_t)r * (S_FAN * D_FEAT) + c;
  float s = 0.f;
#pragma unroll
  for (int i = 0; i < S_FAN; ++i) s += p[i * D_FEAT];
  out[idx] = s * (1.0f / S_FAN);
}

// ---------------------------------------------------------------------------
// Fused concat-GEMM:  out[N][128] = [A1 | A2] @ W   (W: [256][128], given as
// bf16 transposed Wt[n][k]).  Implemented as A1@W_top + A2@W_bot so the
// concat is never materialized.
//
// The entire 64KB bf16 weight matrix is staged ONCE per block into LDS,
// pre-swizzled into the exact per-lane WMMA B-fragment layout:
//   frag(kc,j): lane L holds 16 bf16 = 32 contiguous bytes at
//               sW + ((kc*8 + j)*32 + L)*16 elems
//   where lane L = n_local + 16*h holds column n = j*16 + n_local,
//   chunk0 = K{kc*32 + h*8 .. +7}, chunk1 = K{kc*32 + 16 + h*8 .. +7}
// (16-bit 32x16 B layout, ISA 7.12.2, mirrored from the A map).
// B fragments then come from ds_load_b128 instead of repeated L2 hits.
//
// 4 waves per block; each wave computes a 16-row x 128-col strip with
// v_wmma_f32_16x16x32_bf16 (f32 accumulate), K = 256 over the two halves.
// ---------------------------------------------------------------------------
__global__ void __launch_bounds__(128)
gemm_agg_kernel(const float* __restrict__ A1, const float* __restrict__ A2,
                const __bf16* __restrict__ Wt, float* __restrict__ out,
                int doRelu) {
  __shared__ __align__(32) __bf16 sW[64 * 32 * 16];  // 64 frags * 1KB = 64KB

  const int lane = threadIdx.x & 31;
  const int wave = threadIdx.x >> 5;
  const int r = lane & 15;   // A-row / C-col inside the 16x16 tile
  const int h = lane >> 4;   // lane half
  const size_t rowBase = (size_t)blockIdx.x * 64 + (size_t)wave * 16;
  const size_t aRow = rowBase + (size_t)r;

  // ---- cooperative swizzled weight staging: 4096 x 16B ----
  for (int i = threadIdx.x; i < 4096; i += 128) {
    int f = i >> 6;                 // frag index = kc*8 + j
    int p = i & 63;                 // position within frag
    int l = p >> 1;                 // lane the data belongs to
    int c = p & 1;                  // 16B chunk (0: K+0, 1: K+16)
    int n = (f & 7) * 16 + (l & 15);
    int kb = (f >> 3) * 32 + c * 16 + (l >> 4) * 8;
    ((uint4*)sW)[i] = *(const uint4*)(Wt + (size_t)n * 256 + kb);
  }
  __syncthreads();

  v8f acc[8];
#pragma unroll
  for (int j = 0; j < 8; ++j) acc[j] = (v8f){};

#pragma unroll 2
  for (int kc = 0; kc < 8; ++kc) {           // K = 256 in chunks of 32
    const float* Ap = (kc < 4) ? (A1 + aRow * D_FEAT + kc * 32)
                               : (A2 + aRow * D_FEAT + (kc - 4) * 32);
    float4 f0 = *(const float4*)(Ap + h * 8);
    float4 f1 = *(const float4*)(Ap + h * 8 + 4);
    float4 f2 = *(const float4*)(Ap + 16 + h * 8);
    float4 f3 = *(const float4*)(Ap + 16 + h * 8 + 4);
    v16bf a;
    a[0]  = (__bf16)f0.x; a[1]  = (__bf16)f0.y; a[2]  = (__bf16)f0.z; a[3]  = (__bf16)f0.w;
    a[4]  = (__bf16)f1.x; a[5]  = (__bf16)f1.y; a[6]  = (__bf16)f1.z; a[7]  = (__bf16)f1.w;
    a[8]  = (__bf16)f2.x; a[9]  = (__bf16)f2.y; a[10] = (__bf16)f2.z; a[11] = (__bf16)f2.w;
    a[12] = (__bf16)f3.x; a[13] = (__bf16)f3.y; a[14] = (__bf16)f3.z; a[15] = (__bf16)f3.w;

#pragma unroll
    for (int j = 0; j < 8; ++j) {            // 8 column tiles = 128 cols
      v16bf b = *(const v16bf*)(sW + (size_t)((kc * 8 + j) * 32 + lane) * 16);
      acc[j] = __builtin_amdgcn_wmma_f32_16x16x32_bf16(
          false, a, false, b, (short)0, acc[j], false, false);
    }
  }

#pragma unroll
  for (int j = 0; j < 8; ++j) {
#pragma unroll
    for (int i = 0; i < 8; ++i) {
      float v = acc[j][i];
      if (doRelu) v = fmaxf(v, 0.0f);
      out[(rowBase + (size_t)h * 8 + i) * D_FEAT + j * 16 + r] = v;
    }
  }
}

// ---------------------------------------------------------------------------
// Tiny MLP tail: h1[1024][128] -> relu(@W2[128][64]) -> relu(@W3[64][8])
// -> @W4[8][2] -> softmax.  One wave per row; LDS for cross-lane handoff.
// ---------------------------------------------------------------------------
__global__ void __launch_bounds__(256)
mlp_tail_kernel(const float* __restrict__ h1, const float* __restrict__ W2,
                const float* __restrict__ W3, const float* __restrict__ W4,
                float* __restrict__ out) {
  __shared__ float sh2[8][64];
  __shared__ float sh3[8][8];
  const int wave = threadIdx.x >> 5;
  const int lane = threadIdx.x & 31;
  const int row = blockIdx.x * 8 + wave;
  const float* x = h1 + (size_t)row * 128;

  // layer: 128 -> 64 (each lane computes 2 output columns)
  float s0 = 0.f, s1 = 0.f;
#pragma unroll 4
  for (int k = 0; k < 128; ++k) {
    float xv = x[k];
    s0 += xv * W2[k * 64 + lane];
    s1 += xv * W2[k * 64 + lane + 32];
  }
  sh2[wave][lane]      = fmaxf(s0, 0.f);
  sh2[wave][lane + 32] = fmaxf(s1, 0.f);
  __syncthreads();

  // layer: 64 -> 8
  if (lane < 8) {
    float s = 0.f;
#pragma unroll
    for (int k = 0; k < 64; ++k) s += sh2[wave][k] * W3[k * 8 + lane];
    sh3[wave][lane] = fmaxf(s, 0.f);
  }
  __syncthreads();

  // layer: 8 -> 2 + softmax
  if (lane == 0) {
    float l0 = 0.f, l1 = 0.f;
#pragma unroll
    for (int k = 0; k < 8; ++k) {
      float v = sh3[wave][k];
      l0 += v * W4[k * 2 + 0];
      l1 += v * W4[k * 2 + 1];
    }
    float m  = fmaxf(l0, l1);
    float e0 = __expf(l0 - m), e1 = __expf(l1 - m);
    float inv = 1.f / (e0 + e1);
    out[row * 2 + 0] = e0 * inv;
    out[row * 2 + 1] = e1 * inv;
  }
}

// ---------------------------------------------------------------------------
// Orchestration.  Workspace layout (~28 MB, buffers reused across stages):
//   WT2, WT1 : bf16 transposed weights (64 KB each)
//   buf0,buf1: [25600][128] f32 ping-pong for hop-2 means / hidden states
//   small*   : [1024][128] f32 stage buffers
// ---------------------------------------------------------------------------
extern "C" void kernel_launch(void* const* d_in, const int* in_sizes, int n_in,
                              void* d_out, int out_size, void* d_ws, size_t ws_size,
                              hipStream_t stream) {
  (void)in_sizes; (void)n_in; (void)out_size; (void)ws_size;
  const float* src         = (const float*)d_in[0];
  const float* src_neg     = (const float*)d_in[1];
  const float* src_neg_neg = (const float*)d_in[2];
  const float* dst         = (const float*)d_in[3];
  const float* dst_neg     = (const float*)d_in[4];
  const float* dst_neg_neg = (const float*)d_in[5];
  const float* w2          = (const float*)d_in[6];
  const float* W1          = (const float*)d_in[7];
  const float* W2          = (const float*)d_in[8];
  const float* W3          = (const float*)d_in[9];
  const float* W4          = (const float*)d_in[10];
  float* out = (float*)d_out;

  const int B  = 1024;
  const int BS = B * S_FAN;                    // 25600
  const size_t BIG   = (size_t)BS * D_FEAT * sizeof(float);  // 13,107,200 B
  const size_t SMALL = (size_t)B  * D_FEAT * sizeof(float);  //    524,288 B

  uint8_t* w = (uint8_t*)d_ws;
  __bf16* WT2   = (__bf16*)(w);
  __bf16* WT1   = (__bf16*)(w + 65536);
  float*  buf0  = (float*)(w + 131072);
  float*  buf1  = (float*)(w + 131072 + BIG);
  uint8_t* sm   = w + 131072 + 2 * BIG;
  float*  meanS1 = (float*)(sm);
  float*  meanD1 = (float*)(sm + SMALL);
  float*  srcO   = (float*)(sm + 2 * SMALL);
  float*  dstO   = (float*)(sm + 3 * SMALL);
  float*  h1     = (float*)(sm + 4 * SMALL);

  // weights -> bf16, transposed (L2-resident afterwards)
  wconv_kernel<<<128, 256, 0, stream>>>(w2, WT2);
  wconv_kernel<<<128, 256, 0, stream>>>(W1, WT1);

  // ---- src side, hop 2 ----
  mean_kernel<<<(BS * D_FEAT) / 256, 256, 0, stream>>>(src_neg_neg, buf0, BS);
  gemm_agg_kernel<<<BS / 64, 128, 0, stream>>>(src_neg, buf0, WT2, buf1, 0); // src_h
  mean_kernel<<<(B * D_FEAT) / 256, 256, 0, stream>>>(buf1, meanS1, B);

  // ---- dst side, hop 2 (reuse buf0/buf1) ----
  mean_kernel<<<(BS * D_FEAT) / 256, 256, 0, stream>>>(dst_neg_neg, buf0, BS);
  gemm_agg_kernel<<<BS / 64, 128, 0, stream>>>(dst_neg, buf0, WT2, buf1, 0); // dst_h
  mean_kernel<<<(B * D_FEAT) / 256, 256, 0, stream>>>(buf1, meanD1, B);

  // ---- hop 1 ----
  gemm_agg_kernel<<<B / 64, 128, 0, stream>>>(src, meanS1, WT2, srcO, 0);
  gemm_agg_kernel<<<B / 64, 128, 0, stream>>>(dst, meanD1, WT2, dstO, 0);

  // ---- MLP layer 1: relu([srcO|dstO] @ W1) -- same fused-concat GEMM ----
  gemm_agg_kernel<<<B / 64, 128, 0, stream>>>(srcO, dstO, WT1, h1, 1);

  // ---- MLP tail + softmax ----
  mlp_tail_kernel<<<B / 8, 256, 0, stream>>>(h1, W2, W3, W4, out);
}
